// TeacherForcingMusicLSTM_45432164057811
// MI455X (gfx1250) — matrix operand order repaired and checked by reading, not compiled, not run on hardware
//
#include <hip/hip_runtime.h>
#include <hip/hip_bf16.h>

#define BATCH 64
#define EMB   512
#define HID   1024
#define TXLEN 256
#define VOCAB 2048

// LDS row stride for staged weight tiles: 128 bf16 + 8 pad => 272B rows,
// so lanes reading 16 different rows land on distinct bank groups.
#define LDSTRIDE 136

typedef __attribute__((ext_vector_type(16))) __bf16 v16bf;
typedef __attribute__((ext_vector_type(8)))  float  v8f;

// 128-bit chunk type matching the async builtin's pointee (v4i).
typedef int v4i_ __attribute__((vector_size(16)));
// Explicit address-space-qualified types so LDS accesses lower to ds_* ops
// (a generic pointer select would fall back to flat_load).
typedef __attribute__((address_space(3))) unsigned short us3;
typedef __attribute__((address_space(3))) v4i_ l_v4i;
typedef __attribute__((address_space(1))) v4i_ g_v4i;

union FragU { v4i_ u[2]; v16bf v; };

__device__ __forceinline__ g_v4i* to_global(const void* p) {
    return (g_v4i*)(uintptr_t)p;
}
__device__ __forceinline__ us3* to_lds_us(void* p) {
    // generic LDS address: low 32 bits are the LDS offset
    return (us3*)(unsigned int)(uintptr_t)p;
}

#if defined(__gfx1250__) && __has_builtin(__builtin_amdgcn_global_load_async_to_lds_b128)
#define HAS_ASYNC_LDS 1
#else
#define HAS_ASYNC_LDS 0
#endif

#if HAS_ASYNC_LDS
#  if __has_builtin(__builtin_amdgcn_s_wait_asynccnt)
#    define WAIT_ASYNC(n) __builtin_amdgcn_s_wait_asynccnt(n)
#  else
#    define WAIT_ASYNC(n) asm volatile("s_wait_asynccnt %0" ::"i"(n) : "memory")
#  endif
#else
#define WAIT_ASYNC(n) ((void)0)
#endif

// Scheduling-group hints: 0x100 = DS read, 0x008 = MFMA/WMMA.
#if defined(__gfx1250__) && __has_builtin(__builtin_amdgcn_sched_group_barrier)
#define SGB(mask, size, id) __builtin_amdgcn_sched_group_barrier(mask, size, id)
#else
#define SGB(mask, size, id) ((void)0)
#endif

__device__ __forceinline__ unsigned short f32_bf16(float f) {
    unsigned int u = __float_as_uint(f);
    u += 0x7FFFu + ((u >> 16) & 1u);   // round-to-nearest-even
    return (unsigned short)(u >> 16);
}

// A fragment 16x32 bf16 from row-major A[row][k] (activations, per-wave unique).
// lanes 0-15: M=lane, K = k0+0..7 / k0+16..23 ; lanes 16-31: K = k0+8..15 / +24..31.
__device__ __forceinline__ v16bf load_a_frag(const unsigned short* A, int lda,
                                             int row0, int k0, int lane) {
    const int r  = lane & 15;
    const int ks = lane >> 4;
    const v4i_* p = (const v4i_*)(A + (size_t)(row0 + r) * lda + k0 + ks * 8);
    FragU f;
    f.u[0] = p[0];
    f.u[1] = p[2];
    return f.v;
}

// B fragment 32x16 bf16 from the LDS-staged weight slice (row-major [row][128]+pad).
// lanes 0-15: N=rowbase+lane, K=kk+0..7 / kk+16..23 ; lanes 16-31: K=kk+8..15 / kk+24..31.
__device__ __forceinline__ v16bf lds_b_frag(const us3* buf,
                                            int rowbase, int kk, int lane) {
    const int n  = lane & 15;
    const int ks = lane >> 4;
    const l_v4i* p = (const l_v4i*)(buf + (rowbase + n) * LDSTRIDE + kk + ks * 8);
    FragU f;
    f.u[0] = p[0];
    f.u[1] = p[2];
    return f.v;
}

__device__ __forceinline__ v8f wmma_bf16(v16bf a, v16bf b, v8f c) {
    return __builtin_amdgcn_wmma_f32_16x16x32_bf16(
        false, a, false, b, (short)0, c, false, false);
}

__device__ __forceinline__ float sigmoidf_(float x) {
    return 1.0f / (1.0f + __expf(-x));
}

// Stage one K-chunk (128 cols) of a weight slice into LDS.
// ROWS = 64 (4 gates x 16 hidden cols) or 16 (vocab tile). PT = ROWS/8
// 16-byte segments per thread (128 threads). Async copy when available.
template <int PT>
__device__ __forceinline__ void stage_chunk(const unsigned short* __restrict__ W,
                                            int ldb, int jbase, int gstride, int k0,
                                            us3* dst_base, int tid) {
#pragma unroll
    for (int it = 0; it < PT; ++it) {
        const int s  = tid + it * 128;
        const int r  = s >> 4;            // LDS row
        const int so = (s & 15) * 8;      // ushort offset within row (16B segs)
        const int g  = r >> 4;
        const int jl = r & 15;
        const unsigned short* src =
            W + (size_t)(g * gstride + jbase + jl) * ldb + k0 + so;
        us3* dst = dst_base + r * LDSTRIDE + so;
#if HAS_ASYNC_LDS
        __builtin_amdgcn_global_load_async_to_lds_b128(to_global(src), (l_v4i*)dst, 0, 0);
#else
        *(l_v4i*)dst = *(const v4i_*)src;
#endif
    }
}

// ---------------- prep kernels ----------------

__global__ void cvt_bf16_kernel(const float* __restrict__ src,
                                unsigned short* __restrict__ dst, int n) {
    int i = blockIdx.x * blockDim.x + threadIdx.x;
    if (i < n) dst[i] = f32_bf16(src[i]);
}

__global__ void bias_sum_kernel(const float* __restrict__ a,
                                const float* __restrict__ b,
                                float* __restrict__ o, int n) {
    int i = blockIdx.x * blockDim.x + threadIdx.x;
    if (i < n) o[i] = a[i] + b[i];
}

__global__ void zero_kernel(float* __restrict__ p, int n) {
    int i = blockIdx.x * blockDim.x + threadIdx.x;
    if (i < n) p[i] = 0.0f;
}

// xe[t][b][e] = bf16(emb[x[b][t]][e])
__global__ void embed_kernel(const int* __restrict__ x,
                             const float* __restrict__ emb,
                             unsigned short* __restrict__ xe) {
    int i = blockIdx.x * blockDim.x + threadIdx.x;
    if (i >= TXLEN * BATCH * EMB) return;
    int e  = i % EMB;
    int tb = i / EMB;
    int b  = tb % BATCH;
    int t  = tb / BATCH;
    int tok = x[b * TXLEN + t];
    xe[i] = f32_bf16(emb[(size_t)tok * EMB + e]);
}

// ---------------- staged 4-gate GEMM phase ----------------
// acc[g] += A[64,KTOT] @ W[g*HID + j0 .. +15][0..KTOT]^T,
// weights double-buffered through LDS in 128-deep K chunks (even chunk count).
// B fragments are software-pipelined; sched_group_barrier pins the issue order
// to [8 DS reads][12 x (1 WMMA, 2 DS reads)][4 WMMA] so each WMMA overlaps the
// fetch of a fragment 4 positions ahead instead of stalling on s_wait_dscnt 0.
__device__ __forceinline__ void lstm_gemm_phase(
        const unsigned short* __restrict__ A, int lda,
        const unsigned short* __restrict__ W, int ldb, int KTOT,
        int row0, int j0,
        us3* lds0, int bufstride,
        v8f& a0, v8f& a1, v8f& a2, v8f& a3,
        int lane, int tid) {
    const int nch = KTOT >> 7;
    stage_chunk<8>(W, ldb, j0, HID, 0, lds0, tid);
    for (int ci = 0; ci < nch; ++ci) {
        us3* cur = lds0 + (ci & 1) * bufstride;
        __syncthreads();                       // prior readers of next buffer done
        if (ci + 1 < nch) {
            stage_chunk<8>(W, ldb, j0, HID, (ci + 1) << 7,
                           lds0 + ((ci + 1) & 1) * bufstride, tid);
            WAIT_ASYNC(8);                     // chunk ci complete; ci+1 in flight
        } else {
            WAIT_ASYNC(0);
        }
        __syncthreads();                       // all waves staged chunk ci
        const int kb = ci << 7;

        v16bf bc0 = lds_b_frag(cur, 0,  0, lane);
        v16bf bc1 = lds_b_frag(cur, 16, 0, lane);
        v16bf bc2 = lds_b_frag(cur, 32, 0, lane);
        v16bf bc3 = lds_b_frag(cur, 48, 0, lane);
#pragma unroll
        for (int kk = 0; kk < 4; ++kk) {
            v16bf a = load_a_frag(A, lda, row0, kb + kk * 32, lane);
            v16bf bn0, bn1, bn2, bn3;
            if (kk < 3) {
                bn0 = lds_b_frag(cur, 0,  (kk + 1) * 32, lane);
                bn1 = lds_b_frag(cur, 16, (kk + 1) * 32, lane);
                bn2 = lds_b_frag(cur, 32, (kk + 1) * 32, lane);
                bn3 = lds_b_frag(cur, 48, (kk + 1) * 32, lane);
            }
            a0 = wmma_bf16(a, bc0, a0);
            a1 = wmma_bf16(a, bc1, a1);
            a2 = wmma_bf16(a, bc2, a2);
            a3 = wmma_bf16(a, bc3, a3);
            if (kk < 3) { bc0 = bn0; bc1 = bn1; bc2 = bn2; bc3 = bn3; }
        }
        // Pin issue order: 8 DS reads, 12 x (1 WMMA + 2 DS reads), 4 WMMA.
        SGB(0x100, 8, 0);
#pragma unroll
        for (int i = 0; i < 12; ++i) {
            SGB(0x008, 1, 0);
            SGB(0x100, 2, 0);
        }
        SGB(0x008, 4, 0);
    }
}

// ---------------- LSTM cell: gates GEMM + fused nonlinearity/state update ----
// Grid: 64 blocks x 128 threads (4 waves). All 4 waves of a block share the
// hidden-col tile j0 = blockIdx*16 (weights staged once in LDS); wave = row tile.
__global__ __launch_bounds__(128) void lstm_cell_kernel(
        const unsigned short* __restrict__ A1, int K1,
        const unsigned short* __restrict__ Wih,
        const unsigned short* __restrict__ h_in,
        const unsigned short* __restrict__ Whh,
        const float* __restrict__ bias,
        float* __restrict__ c,
        unsigned short* __restrict__ h_out) {
    __shared__ unsigned short lb[2][64 * LDSTRIDE];
    const int tid  = threadIdx.x;
    const int lane = tid & 31;
    const int row0 = (tid >> 5) * 16;           // batch-row tile (per wave)
    const int j0   = blockIdx.x * 16;           // hidden-col tile (per block)
    us3* lds0 = to_lds_us(&lb[0][0]);

    v8f acc_i = {}, acc_f = {}, acc_g = {}, acc_o = {};

    lstm_gemm_phase(A1,   K1,  Wih, K1,  K1,  row0, j0, lds0, 64 * LDSTRIDE,
                    acc_i, acc_f, acc_g, acc_o, lane, tid);
    lstm_gemm_phase(h_in, HID, Whh, HID, HID, row0, j0, lds0, 64 * LDSTRIDE,
                    acc_i, acc_f, acc_g, acc_o, lane, tid);

    // C/D layout: lanes 0-15 -> N=lane, M=row0+v ; lanes 16-31 -> M=row0+8+v.
    const int n  = lane & 15;
    const int mb = (lane >> 4) * 8;
    const int j  = j0 + n;
    const float bi = bias[j];
    const float bf = bias[HID + j];
    const float bg = bias[2 * HID + j];
    const float bo = bias[3 * HID + j];

#pragma unroll
    for (int v = 0; v < 8; ++v) {
        const int m = row0 + mb + v;
        const size_t idx = (size_t)m * HID + j;
        float ig = sigmoidf_(acc_i[v] + bi);
        float fg = sigmoidf_(acc_f[v] + bf);
        float gg = tanhf(acc_g[v] + bg);
        float og = sigmoidf_(acc_o[v] + bo);
        float cv = fg * c[idx] + ig * gg;
        c[idx] = cv;
        h_out[idx] = f32_bf16(og * tanhf(cv));
    }
}

// ---------------- output projection: logits = h1 @ Wout^T + b_out ------------
// Grid: 128 blocks x 128 threads. 4 waves share vocab tile n0 = blockIdx*16.
__global__ __launch_bounds__(128) void out_proj_kernel(
        const unsigned short* __restrict__ h,
        const unsigned short* __restrict__ Wout,
        const float* __restrict__ b_out,
        float* __restrict__ out) {
    __shared__ unsigned short lb[2][16 * LDSTRIDE];
    const int tid  = threadIdx.x;
    const int lane = tid & 31;
    const int row0 = (tid >> 5) * 16;
    const int n0   = blockIdx.x * 16;
    us3* lds0 = to_lds_us(&lb[0][0]);

    v8f acc = {};
    const int nch = HID >> 7;                   // 8 chunks (even)
    stage_chunk<2>(Wout, HID, n0, 0, 0, lds0, tid);
    for (int ci = 0; ci < nch; ++ci) {
        us3* cur = lds0 + (ci & 1) * (16 * LDSTRIDE);
        __syncthreads();
        if (ci + 1 < nch) {
            stage_chunk<2>(Wout, HID, n0, 0, (ci + 1) << 7,
                           lds0 + ((ci + 1) & 1) * (16 * LDSTRIDE), tid);
            WAIT_ASYNC(2);
        } else {
            WAIT_ASYNC(0);
        }
        __syncthreads();
        const int kb = ci << 7;

        v16bf bc = lds_b_frag(cur, 0, 0, lane);
#pragma unroll
        for (int kk = 0; kk < 4; ++kk) {
            v16bf a = load_a_frag(h, HID, row0, kb + kk * 32, lane);
            v16bf bn;
            if (kk < 3) bn = lds_b_frag(cur, 0, (kk + 1) * 32, lane);
            acc = wmma_bf16(a, bc, acc);
            if (kk < 3) bc = bn;
        }
        // Pin issue order: 2 DS reads, 3 x (1 WMMA + 2 DS reads), 1 WMMA.
        SGB(0x100, 2, 0);
#pragma unroll
        for (int i = 0; i < 3; ++i) {
            SGB(0x008, 1, 0);
            SGB(0x100, 2, 0);
        }
        SGB(0x008, 1, 0);
    }

    const int n  = n0 + (lane & 15);
    const int mb = (lane >> 4) * 8;
    const float bn = b_out[n];
#pragma unroll
    for (int v = 0; v < 8; ++v) {
        out[(size_t)(row0 + mb + v) * VOCAB + n] = acc[v] + bn;
    }
}

// ---------------- host side ----------------

extern "C" void kernel_launch(void* const* d_in, const int* in_sizes, int n_in,
                              void* d_out, int out_size, void* d_ws, size_t ws_size,
                              hipStream_t stream) {
    (void)in_sizes; (void)n_in; (void)out_size; (void)ws_size;

    const int*   x    = (const int*)d_in[0];
    const float* emb  = (const float*)d_in[1];
    const float* Wih1 = (const float*)d_in[2];
    const float* Whh1 = (const float*)d_in[3];
    const float* bih1 = (const float*)d_in[4];
    const float* bhh1 = (const float*)d_in[5];
    const float* Wih2 = (const float*)d_in[6];
    const float* Whh2 = (const float*)d_in[7];
    const float* bih2 = (const float*)d_in[8];
    const float* bhh2 = (const float*)d_in[9];
    const float* Wout = (const float*)d_in[10];
    const float* bout = (const float*)d_in[11];
    float* out = (float*)d_out;

    char* ws = (char*)d_ws;
    size_t off = 0;
    auto alloc = [&](size_t bytes) -> void* {
        void* p = ws + off;
        off += (bytes + 255) & ~(size_t)255;
        return p;
    };

    unsigned short* wih1b = (unsigned short*)alloc((size_t)4 * HID * EMB * 2);
    unsigned short* whh1b = (unsigned short*)alloc((size_t)4 * HID * HID * 2);
    unsigned short* wih2b = (unsigned short*)alloc((size_t)4 * HID * HID * 2);
    unsigned short* whh2b = (unsigned short*)alloc((size_t)4 * HID * HID * 2);
    unsigned short* woutb = (unsigned short*)alloc((size_t)VOCAB * HID * 2);
    unsigned short* xe    = (unsigned short*)alloc((size_t)TXLEN * BATCH * EMB * 2);
    float* bias1 = (float*)alloc(4 * HID * sizeof(float));
    float* bias2 = (float*)alloc(4 * HID * sizeof(float));
    // state block (zeroed each call): h0[2] bf16, h1[2] bf16, c0, c1 f32
    char* sblock = (char*)alloc((size_t)4 * BATCH * HID * 2 + (size_t)2 * BATCH * HID * 4);
    unsigned short* h0buf[2] = { (unsigned short*)sblock,
                                 (unsigned short*)sblock + (size_t)BATCH * HID };
    unsigned short* h1buf[2] = { (unsigned short*)sblock + (size_t)2 * BATCH * HID,
                                 (unsigned short*)sblock + (size_t)3 * BATCH * HID };
    float* c0 = (float*)(sblock + (size_t)4 * BATCH * HID * 2);
    float* c1 = c0 + (size_t)BATCH * HID;

    const int T = 256;
    // --- prep: weight/bias conversion, state zeroing, embedding gather ---
    cvt_bf16_kernel<<<(4 * HID * EMB + T - 1) / T, T, 0, stream>>>(Wih1, wih1b, 4 * HID * EMB);
    cvt_bf16_kernel<<<(4 * HID * HID + T - 1) / T, T, 0, stream>>>(Whh1, whh1b, 4 * HID * HID);
    cvt_bf16_kernel<<<(4 * HID * HID + T - 1) / T, T, 0, stream>>>(Wih2, wih2b, 4 * HID * HID);
    cvt_bf16_kernel<<<(4 * HID * HID + T - 1) / T, T, 0, stream>>>(Whh2, whh2b, 4 * HID * HID);
    cvt_bf16_kernel<<<(VOCAB * HID + T - 1) / T, T, 0, stream>>>(Wout, woutb, VOCAB * HID);
    bias_sum_kernel<<<(4 * HID + T - 1) / T, T, 0, stream>>>(bih1, bhh1, bias1, 4 * HID);
    bias_sum_kernel<<<(4 * HID + T - 1) / T, T, 0, stream>>>(bih2, bhh2, bias2, 4 * HID);
    {
        int nz = (4 * BATCH * HID * 2 + 2 * BATCH * HID * 4) / 4;
        zero_kernel<<<(nz + T - 1) / T, T, 0, stream>>>((float*)sblock, nz);
    }
    embed_kernel<<<(TXLEN * BATCH * EMB + T - 1) / T, T, 0, stream>>>(x, emb, xe);

    // --- 256 timesteps: L1 cell -> L2 cell -> logits (stream order = deps) ---
    for (int t = 0; t < TXLEN; ++t) {
        const int cur = t & 1, nxt = cur ^ 1;
        lstm_cell_kernel<<<64, 128, 0, stream>>>(
            xe + (size_t)t * BATCH * EMB, EMB, wih1b,
            h0buf[cur], whh1b, bias1, c0, h0buf[nxt]);
        lstm_cell_kernel<<<64, 128, 0, stream>>>(
            h0buf[nxt], HID, wih2b,
            h1buf[cur], whh2b, bias2, c1, h1buf[nxt]);
        out_proj_kernel<<<128, 128, 0, stream>>>(
            h1buf[nxt], woutb, bout, out + (size_t)t * BATCH * VOCAB);
    }
}
